// EdgePredictor_52493090292432
// MI455X (gfx1250) — matrix operand order repaired and checked by reading, not compiled, not run on hardware
//
#include <hip/hip_runtime.h>
#include <hip/hip_bf16.h>

typedef __attribute__((ext_vector_type(16))) _Float16 v16h;
typedef __attribute__((ext_vector_type(8)))  float    v8f;

#define NN  200000
#define NG  100000
#define NU  100000
#define EE  3200000
#define CC  1000000

// ---------------- zero scratch (agg buffers) ----------------
__global__ void k_zero(float* __restrict__ p, int n) {
    int i = blockIdx.x * blockDim.x + threadIdx.x;
    if (i < n) p[i] = 0.f;
}

// ---------------- x_unconnected = x[NG:] @ wu + bu ----------------
__global__ void k_xu(const float* __restrict__ x, const float* __restrict__ wu,
                     const float* __restrict__ bu, float* __restrict__ xu) {
    int t = blockIdx.x * blockDim.x + threadIdx.x;
    if (t >= NU * 16) return;
    int i = t >> 4, j = t & 15;
    float x0 = x[(size_t)(NG + i) * 2 + 0];
    float x1 = x[(size_t)(NG + i) * 2 + 1];
    xu[t] = fmaf(x0, wu[j], fmaf(x1, wu[16 + j], bu[j]));
}

// ---------------- layer-1 scatter: agg1[dst] += x[src] (d=2) ----------------
__global__ void k_scat1(const float* __restrict__ x, const int* __restrict__ edges,
                        float* __restrict__ agg1) {
    int e = blockIdx.x * blockDim.x + threadIdx.x;
    if (e >= EE) return;
    int s = edges[e], d = edges[EE + e];
    atomicAdd(&agg1[(size_t)d * 2 + 0], x[(size_t)s * 2 + 0]);
    atomicAdd(&agg1[(size_t)d * 2 + 1], x[(size_t)s * 2 + 1]);
}

// ---------------- h1 = relu(agg1@w1_rel + x@w1_root + b1)  (NG x 8) ----------------
__global__ void k_h1(const float* __restrict__ x, const float* __restrict__ agg1,
                     const float* __restrict__ w1r, const float* __restrict__ w1o,
                     const float* __restrict__ b1, float* __restrict__ h1) {
    int t = blockIdx.x * blockDim.x + threadIdx.x;
    if (t >= NG * 8) return;
    int i = t >> 3, j = t & 7;
    float a0 = agg1[(size_t)i * 2 + 0], a1 = agg1[(size_t)i * 2 + 1];
    float x0 = x[(size_t)i * 2 + 0],    x1 = x[(size_t)i * 2 + 1];
    float v = b1[j];
    v = fmaf(a0, w1r[j], v); v = fmaf(a1, w1r[8 + j], v);
    v = fmaf(x0, w1o[j], v); v = fmaf(x1, w1o[8 + j], v);
    h1[t] = fmaxf(v, 0.f);
}

// ---------------- layer-2 scatter: agg2[dst] += h1[src] (d=8) ----------------
__global__ void k_scat2(const float* __restrict__ h1, const int* __restrict__ edges,
                        float* __restrict__ agg2) {
    int e = blockIdx.x * blockDim.x + threadIdx.x;
    if (e >= EE) return;
    int s = edges[e], d = edges[EE + e];
    const float* hs = h1 + (size_t)s * 8;
    float* ad = agg2 + (size_t)d * 8;
#pragma unroll
    for (int j = 0; j < 8; ++j) atomicAdd(ad + j, hs[j]);
}

// ---------------- h2 = relu(agg2@w2_rel + h1@w2_root + b2)  (NG x 16) ----------------
__global__ void k_h2(const float* __restrict__ h1, const float* __restrict__ agg2,
                     const float* __restrict__ w2r, const float* __restrict__ w2o,
                     const float* __restrict__ b2, float* __restrict__ h2) {
    int t = blockIdx.x * blockDim.x + threadIdx.x;
    if (t >= NG * 16) return;
    int i = t >> 4, j = t & 15;
    float v = b2[j];
#pragma unroll
    for (int k = 0; k < 8; ++k) {
        v = fmaf(agg2[(size_t)i * 8 + k], w2r[k * 16 + j], v);
        v = fmaf(h1[(size_t)i * 8 + k],   w2o[k * 16 + j], v);
    }
    h2[t] = fmaxf(v, 0.f);
}

// ---------------- candidate MLP with WMMA ----------------
// Per wave: 16 candidates. A = ec tile [16x32] f16 (gathered as float4, cvt_pk to f16);
// B = wa staged in LDS pre-swizzled to per-lane fragment layout -> 2x ds_load_b128/tile;
// C preloaded with ba; relu; 64->1 layer via per-lane wb dot + 16-lane xor reduction.
__global__ __launch_bounds__(128) void k_mlp(
    const float* __restrict__ h2, const float* __restrict__ xu,
    const int* __restrict__ cand,
    const float* __restrict__ wa, const float* __restrict__ ba,
    const float* __restrict__ wb, const float* __restrict__ bb,
    float* __restrict__ out) {
    // sB[(t*32 + lane)*16 + e] = (f16) wa[(hi*8 + e + (e & 8))*64 + t*16 + n],
    // where hi = lane>>4, n = lane&15: each lane's B fragment is 32 contiguous bytes.
    __shared__ __align__(32) _Float16 sB[4 * 32 * 16];
    int tid = threadIdx.x;
    {
        int t  = tid >> 5;          // tile this thread stages
        int hi = (tid >> 4) & 1;
        int n  = tid & 15;
        _Float16* dstp = sB + tid * 16;
        const float* srcp = wa + hi * 8 * 64 + t * 16 + n;
#pragma unroll
        for (int e = 0; e < 16; ++e) dstp[e] = (_Float16)srcp[(e + (e & 8)) * 64];
    }
    __syncthreads();

    int wave = tid >> 5;
    int l    = tid & 31;
    int hi   = l >> 4;      // lane half: K sub-range (A) / K offset (B) / M offset (C,D)
    int n    = l & 15;      // N index for B/C/D; M (row owner) for A
    int tile = blockIdx.x * 4 + wave;     // 62500 tiles exactly -> EXEC all-1s everywhere
    int base = tile * 16;

    // ---- A fragment: row m = n; features 0..15 from h2[c0], 16..31 from xu[c1]
    int ci = base + n;
    int2 cc = ((const int2*)cand)[ci];
    const float4* f0 = (const float4*)(h2 + (size_t)cc.x * 16);   // 64B-aligned rows
    const float4* f1 = (const float4*)(xu + (size_t)cc.y * 16);
    float4 a0 = f0[hi * 2 + 0], a1 = f0[hi * 2 + 1];   // K = hi*8 .. hi*8+7
    float4 a2 = f1[hi * 2 + 0], a3 = f1[hi * 2 + 1];   // K = 16 + hi*8 .. +7
    v16h A;
    A[0]  = (_Float16)a0.x; A[1]  = (_Float16)a0.y; A[2]  = (_Float16)a0.z; A[3]  = (_Float16)a0.w;
    A[4]  = (_Float16)a1.x; A[5]  = (_Float16)a1.y; A[6]  = (_Float16)a1.z; A[7]  = (_Float16)a1.w;
    A[8]  = (_Float16)a2.x; A[9]  = (_Float16)a2.y; A[10] = (_Float16)a2.z; A[11] = (_Float16)a2.w;
    A[12] = (_Float16)a3.x; A[13] = (_Float16)a3.y; A[14] = (_Float16)a3.z; A[15] = (_Float16)a3.w;

    float p[8];
#pragma unroll
    for (int j = 0; j < 8; ++j) p[j] = 0.f;

    const v16h* Bfrag = (const v16h*)sB;
#pragma unroll
    for (int t = 0; t < 4; ++t) {         // four 16-wide N tiles of the 64-wide hidden layer
        v16h B = Bfrag[t * 32 + l];       // contiguous 32B -> ds_load_b128 x2
        float bias = ba[t * 16 + n];
        v8f acc;
#pragma unroll
        for (int j = 0; j < 8; ++j) acc[j] = bias;
        acc = __builtin_amdgcn_wmma_f32_16x16x32_f16(
            false, A, false, B, (short)0, acc, false, false);
        float w = wb[t * 16 + n];
#pragma unroll
        for (int j = 0; j < 8; ++j) p[j] += fmaxf(acc[j], 0.f) * w;   // relu + 2nd layer
    }

    // reduce over N (16 lanes of this half-wave); offsets 1..8 never cross the half
#pragma unroll
    for (int j = 0; j < 8; ++j) {
#pragma unroll
        for (int off = 8; off; off >>= 1) p[j] += __shfl_xor(p[j], off, 32);
    }
    float bb0 = bb[0];
#pragma unroll
    for (int j = 0; j < 8; ++j)
        if (n == j) out[base + hi * 8 + j] = p[j] + bb0;   // candidate M = hi*8 + j
}

extern "C" void kernel_launch(void* const* d_in, const int* in_sizes, int n_in,
                              void* d_out, int out_size, void* d_ws, size_t ws_size,
                              hipStream_t stream) {
    const float* x      = (const float*)d_in[0];
    // d_in[1] = mask: structurally arange(N) < NG, not needed
    const int*   cand   = (const int*)d_in[2];
    const int*   edges  = (const int*)d_in[3];
    const float* w1_rel = (const float*)d_in[4];
    const float* w1_root= (const float*)d_in[5];
    const float* b1     = (const float*)d_in[6];
    const float* w2_rel = (const float*)d_in[7];
    const float* w2_root= (const float*)d_in[8];
    const float* b2     = (const float*)d_in[9];
    const float* wu     = (const float*)d_in[10];
    const float* bu     = (const float*)d_in[11];
    const float* wa     = (const float*)d_in[12];
    const float* ba     = (const float*)d_in[13];
    const float* wb     = (const float*)d_in[14];
    const float* bb     = (const float*)d_in[15];
    float* out = (float*)d_out;

    float* ws   = (float*)d_ws;
    float* xu   = ws;                       // NU*16
    float* h1   = xu   + (size_t)NU * 16;   // NG*8
    float* h2   = h1   + (size_t)NG * 8;    // NG*16
    float* agg1 = h2   + (size_t)NG * 16;   // NG*2
    float* agg2 = agg1 + (size_t)NG * 2;    // NG*8  (contiguous with agg1)

    k_zero <<<(NG * 10 + 255) / 256, 256, 0, stream>>>(agg1, NG * 10);
    k_xu   <<<(NU * 16 + 255) / 256, 256, 0, stream>>>(x, wu, bu, xu);
    k_scat1<<<(EE + 255) / 256, 256, 0, stream>>>(x, edges, agg1);
    k_h1   <<<(NG * 8 + 255) / 256, 256, 0, stream>>>(x, agg1, w1_rel, w1_root, b1, h1);
    k_scat2<<<(EE + 255) / 256, 256, 0, stream>>>(h1, edges, agg2);
    k_h2   <<<(NG * 16 + 255) / 256, 256, 0, stream>>>(h1, agg2, w2_rel, w2_root, b2, h2);
    k_mlp  <<<CC / 64, 128, 0, stream>>>(h2, xu, cand, wa, ba, wb, bb, out);
}
